// UNET_wave_call_conv_22574348108590
// MI455X (gfx1250) — compile-verified, matrix operand order checked by reading, not run on hardware
//
#include <hip/hip_runtime.h>
#include <hip/hip_bf16.h>
#include <stdint.h>

// ---------------------------------------------------------------------------
// CDNA5 (gfx1250) U-Net forward. Implicit-GEMM 3x3 conv + 2x2 up-conv on
// v_wmma_f32_16x16x32_f16 (wave32). Weight tiles streamed to LDS by the
// Tensor Data Mover (tensor_load_to_lds + s_wait_tensorcnt); im2col B tiles
// gathered branch-free. Double-buffered LDS, b128 fragment loads.
// ---------------------------------------------------------------------------

typedef __attribute__((ext_vector_type(16))) _Float16 v16h;
typedef __attribute__((ext_vector_type(8)))  _Float16 v8h;
typedef __attribute__((ext_vector_type(8)))  float    v8f;

typedef unsigned int u32x4 __attribute__((ext_vector_type(4)));
typedef int          i32x4 __attribute__((ext_vector_type(4)));
typedef int          i32x8 __attribute__((ext_vector_type(8)));

#define EPSF 1e-5f

#if defined(__has_builtin)
#if __has_builtin(__builtin_amdgcn_tensor_load_to_lds) && \
    __has_builtin(__builtin_amdgcn_s_wait_tensorcnt)
#define USE_TDM 1
#endif
#endif

// Per ISA 7.12.2 (16-bit 16x32 fragment): lane holds two contiguous 8-half
// runs: k = base0..base0+7 and k = 16+base0..+7, base0 = (lane&16)?8:0.
static __device__ __forceinline__ v16h frag_load(const _Float16* rowp, int base0) {
  v8h lo = *(const v8h*)(rowp + base0);
  v8h hi = *(const v8h*)(rowp + 16 + base0);
  return __builtin_shufflevector(lo, hi, 0, 1, 2, 3, 4, 5, 6, 7,
                                 8, 9, 10, 11, 12, 13, 14, 15);
}

#ifdef USE_TDM
// TDM: load a 64(row) x 32(col) f16 tile from a row-major [Mp x CKp] matrix
// into LDS, padding each 64B row with 16B (-> 80B LDS row stride).
// D# per ISA 8.3/8.4: group0 = {count,lds_addr,global_addr,type=2},
// group1 = {data_size=2B, pad_en, pad_interval=3 (16 DW), pad_amount=3 (4 DW),
//           tensor_dim0=CKp, tensor_dim1=Mp, tile_dim0=32, tile_dim1=64,
//           tensor_dim0_stride=CKp}.
static __device__ __forceinline__ void tdm_load_A(const _Float16* gsrc,
                                                  unsigned lds_off,
                                                  int CKp, int Mp) {
  unsigned long long ga = (unsigned long long)(uintptr_t)gsrc;
  u32x4 g0;
  g0[0] = 1u;                                   // count=1, gather off
  g0[1] = lds_off;                              // lds_addr
  g0[2] = (unsigned)ga;                         // global_addr[31:0]
  g0[3] = (unsigned)(ga >> 32) | 0x80000000u;   // global_addr[56:32], type=2
  i32x8 g1;
  g1[0] = (1 << 16) | (1 << 20) | (3 << 22) | (3 << 25);
  g1[1] = (int)((CKp & 0xffff) << 16);                       // dim0[15:0]
  g1[2] = (int)(((CKp >> 16) & 0xffff) | ((Mp & 0xffff) << 16));
  g1[3] = (int)(((Mp >> 16) & 0xffff) | (32u << 16));        // tile_dim0=32
  g1[4] = 64;                                                // tile_dim1=64
  g1[5] = CKp;                                               // dim0_stride lo
  g1[6] = 0;
  g1[7] = 0;
  i32x4 z4 = {0, 0, 0, 0};
#if __clang_major__ >= 23
  i32x8 z8 = {0, 0, 0, 0, 0, 0, 0, 0};
  __builtin_amdgcn_tensor_load_to_lds(g0, g1, z4, z4, z8, 0);
#else
  __builtin_amdgcn_tensor_load_to_lds(g0, g1, z4, z4, 0);
#endif
}
#define TDM_DRAIN() __builtin_amdgcn_s_wait_tensorcnt(0)
#else
#define TDM_DRAIN() ((void)0)
#endif

// ---------------------------------------------------------------------------
// Implicit-GEMM 3x3 conv (pad=1) + fused (alpha,beta) affine (+ReLU).
//   A (f16 packed): [Mp x CKp], CKp = 9*Cip, K order = tap*Cip + ci
//   B: branch-free im2col of X, staged transposed in LDS as [N][K]
//   Block: 256 thr = 8 waves; block tile 64(M) x 64(N); wave tile 16x32.
//   FULLC: Cin is a multiple of 32 (no channel-tail guard needed).
// ---------------------------------------------------------------------------
template <bool FULLC>
__global__ __launch_bounds__(256)
void conv3x3_bn_wmma(const _Float16* __restrict__ Apack, int CKp, int Mp,
                     const float* __restrict__ X, float* __restrict__ Y,
                     const float* __restrict__ alpha, const float* __restrict__ beta,
                     int Cin, int H, int W, int O, int relu, int sptLog)
{
  const int HW = H * W;

  __shared__ _Float16 As[2][64][40];   // [buf][M][K]  80B rows (TDM-padded)
  __shared__ _Float16 Bt[2][64][40];   // [buf][N][K]  transposed B tile

  const int tid  = threadIdx.x;
  const int lane = tid & 31;
  const int wave = tid >> 5;
  const int wm   = (wave & 3) << 4;
  const int wn   = (wave >> 2) << 5;

  const int n0 = blockIdx.x * 64;
  const int m0 = blockIdx.y * 64;

#ifdef USE_TDM
  const unsigned ldsA0 = (unsigned)(uintptr_t)&As[0][0][0];
#else
  const int arow = tid >> 2;
  const int acol = (tid & 3) << 3;
  const _Float16* arowp = Apack + (size_t)(m0 + arow) * CKp + acol;
#endif
  const int nc  = tid & 63;
  const int kr0 = (tid >> 6) << 3;

  // loop-invariant n -> (b, oh, ow) decode (all layer N are multiples of 64)
  const int n   = n0 + nc;
  const int b_  = n / HW;
  const int hw  = n - b_ * HW;
  const int oh  = hw / W;
  const int ow  = hw - oh * W;
  const size_t bbase = (size_t)b_ * Cin * HW;

  const int spt    = 1 << sptLog;
  const int nsteps = 9 << sptLog;

  v8f acc0 = {}, acc1 = {};

  auto fillA = [&](int ks, int buf) {
#ifdef USE_TDM
    if (wave == 0)
      tdm_load_A(Apack + (size_t)m0 * CKp + (ks << 5),
                 ldsA0 + (unsigned)buf * 5120u, CKp, Mp);
#else
    const _Float16* src = arowp + (ks << 5);
    *(v8h*)&As[buf][arow][acol] = *(const v8h*)src;
    __builtin_prefetch(src + 32, 0, 1);
#endif
  };

  auto fillB = [&](int ks, int buf) {
    const int tap = ks >> sptLog;
    const int cc  = (ks & (spt - 1)) << 5;
    const int r   = (tap * 11) >> 5;            // tap/3 for tap in 0..8
    const int s   = tap - r * 3;
    const int ih  = oh + r - 1;
    const int iw  = ow + s - 1;
    const bool ok = (ih >= 0) & (ih < H) & (iw >= 0) & (iw < W);
    const int ihC = min(max(ih, 0), H - 1);     // clamp: loads always legal
    const int iwC = min(max(iw, 0), W - 1);
    const float* xp = X + bbase + (size_t)ihC * W + iwC;
    v8h pk;
    if (FULLC) {
      const float* xc = xp + (size_t)(cc + kr0) * HW;
      #pragma unroll
      for (int t = 0; t < 8; ++t) {
        float v = xc[(size_t)t * HW];
        pk[t] = (_Float16)(ok ? v : 0.0f);
      }
    } else {                                    // first layer (Cin=3)
      #pragma unroll
      for (int t = 0; t < 8; ++t) {
        int ci  = cc + kr0 + t;
        int cio = ci < Cin ? ci : Cin - 1;
        float v = xp[(size_t)cio * HW];
        pk[t] = (_Float16)((ok && ci < Cin) ? v : 0.0f);
      }
    }
    *(v8h*)&Bt[buf][nc][kr0] = pk;
  };

  fillA(0, 0);
  fillB(0, 0);
  TDM_DRAIN();
  __syncthreads();

  int cur = 0;
  const int base0 = (lane & 16) ? 8 : 0;
  const int mrow  = wm + (lane & 15);
  const int nrow  = wn + (lane & 15);
  for (int ks = 0; ks < nsteps; ++ks) {
    if (ks + 1 < nsteps) {
      fillA(ks + 1, cur ^ 1);
      fillB(ks + 1, cur ^ 1);
    }
    v16h a  = frag_load(&As[cur][mrow][0], base0);
    v16h b0 = frag_load(&Bt[cur][nrow][0], base0);
    v16h b1 = frag_load(&Bt[cur][nrow + 16][0], base0);
    acc0 = __builtin_amdgcn_wmma_f32_16x16x32_f16(false, a, false, b0,
                                                  (short)0, acc0, false, false);
    acc1 = __builtin_amdgcn_wmma_f32_16x16x32_f16(false, a, false, b1,
                                                  (short)0, acc1, false, false);
    TDM_DRAIN();                         // scalar; no-op for non-issuing waves
    __syncthreads();
    cur ^= 1;
  }

  // epilogue: C layout: VGPR r -> M = r + 8*(lane>=16), N = lane&15
  const int mbase = m0 + wm + ((lane >> 4) << 3);
  const int na = n0 + wn + (lane & 15);
  const int nb = na + 16;
  const int ba  = na / HW;  const int hwa = na - ba * HW;
  const int bb_ = nb / HW;  const int hwb = nb - bb_ * HW;
  #pragma unroll
  for (int r = 0; r < 8; ++r) {
    int m = mbase + r;
    if (m < O) {
      float al = alpha[m], be = beta[m];
      float v0 = acc0[r] * al + be;
      float v1 = acc1[r] * al + be;
      if (relu) { v0 = fmaxf(v0, 0.0f); v1 = fmaxf(v1, 0.0f); }
      Y[((size_t)ba  * O + m) * HW + hwa] = v0;
      Y[((size_t)bb_ * O + m) * HW + hwb] = v1;
    }
  }
}

// ---------------------------------------------------------------------------
// Transposed 2x2 stride-2 conv as 4 WMMA GEMMs (blockIdx.z = rs tap).
// ---------------------------------------------------------------------------
__global__ __launch_bounds__(256)
void upconv_wmma(const _Float16* __restrict__ Apack0, int CKp, int Mp,
                 const float* __restrict__ X, float* __restrict__ Y,
                 const float* __restrict__ bias,
                 int Ci, int H, int W, int O, int Ctot)
{
  const int rs = blockIdx.z;
  const _Float16* Apack = Apack0 + (size_t)rs * Mp * CKp;
  const int HW = H * W;

  __shared__ _Float16 As[2][64][40];
  __shared__ _Float16 Bt[2][64][40];

  const int tid  = threadIdx.x;
  const int lane = tid & 31;
  const int wave = tid >> 5;
  const int wm   = (wave & 3) << 4;
  const int wn   = (wave >> 2) << 5;

  const int n0 = blockIdx.x * 64;
  const int m0 = blockIdx.y * 64;

#ifdef USE_TDM
  const unsigned ldsA0 = (unsigned)(uintptr_t)&As[0][0][0];
#else
  const int arow = tid >> 2;
  const int acol = (tid & 3) << 3;
  const _Float16* arowp = Apack + (size_t)(m0 + arow) * CKp + acol;
#endif
  const int nc  = tid & 63;
  const int kr0 = (tid >> 6) << 3;

  const int n  = n0 + nc;
  const int b_ = n / HW;
  const int hw = n - b_ * HW;
  const size_t xb = (size_t)b_ * Ci * HW + hw;

  const int nsteps = CKp >> 5;

  v8f acc0 = {}, acc1 = {};

  auto fill = [&](int ks, int buf) {
    const int ktile = ks << 5;
#ifdef USE_TDM
    if (wave == 0)
      tdm_load_A(Apack + (size_t)m0 * CKp + ktile,
                 ldsA0 + (unsigned)buf * 5120u, CKp, Mp);
#else
    *(v8h*)&As[buf][arow][acol] = *(const v8h*)(arowp + ktile);
#endif
    const float* xc = X + xb + (size_t)(ktile + kr0) * HW;
    v8h pk;
    #pragma unroll
    for (int t = 0; t < 8; ++t)
      pk[t] = (_Float16)xc[(size_t)t * HW];
    *(v8h*)&Bt[buf][nc][kr0] = pk;
  };

  fill(0, 0);
  TDM_DRAIN();
  __syncthreads();

  int cur = 0;
  const int base0 = (lane & 16) ? 8 : 0;
  const int mrow  = wm + (lane & 15);
  const int nrow  = wn + (lane & 15);
  for (int ks = 0; ks < nsteps; ++ks) {
    if (ks + 1 < nsteps) fill(ks + 1, cur ^ 1);
    v16h a  = frag_load(&As[cur][mrow][0], base0);
    v16h b0 = frag_load(&Bt[cur][nrow][0], base0);
    v16h b1 = frag_load(&Bt[cur][nrow + 16][0], base0);
    acc0 = __builtin_amdgcn_wmma_f32_16x16x32_f16(false, a, false, b0,
                                                  (short)0, acc0, false, false);
    acc1 = __builtin_amdgcn_wmma_f32_16x16x32_f16(false, a, false, b1,
                                                  (short)0, acc1, false, false);
    TDM_DRAIN();
    __syncthreads();
    cur ^= 1;
  }

  const int mbase = m0 + wm + ((lane >> 4) << 3);
  const int na = n0 + wn + (lane & 15);
  const int nb = na + 16;
  const int r_ = rs >> 1, s_ = rs & 1;
  const int ba  = na / HW;  const int hwa = na - ba * HW;
  const int ha  = hwa / W;  const int wa  = hwa - ha * W;
  const int bb_ = nb / HW;  const int hwb = nb - bb_ * HW;
  const int hb  = hwb / W;  const int wb  = hwb - hb * W;
  #pragma unroll
  for (int r = 0; r < 8; ++r) {
    int m = mbase + r;
    if (m < O) {
      float bv = bias[m];
      Y[(((size_t)ba  * Ctot + m) * (2 * H) + 2 * ha + r_) * (2 * W) + 2 * wa + s_] = acc0[r] + bv;
      Y[(((size_t)bb_ * Ctot + m) * (2 * H) + 2 * hb + r_) * (2 * W) + 2 * wb + s_] = acc1[r] + bv;
    }
  }
}

// ---------------------------------------------------------------------------
// Weight packing + BN folding + cheap elementwise kernels
// ---------------------------------------------------------------------------
__global__ void pack_w3x3(const float* __restrict__ w, _Float16* __restrict__ A,
                          int O, int Cin, int Cip, int Mp)
{
  const int CKp = 9 * Cip;
  size_t i = (size_t)blockIdx.x * 256 + threadIdx.x;
  size_t total = (size_t)Mp * CKp;
  if (i >= total) return;
  int m   = (int)(i / CKp);
  int k   = (int)(i - (size_t)m * CKp);
  int tap = k / Cip;
  int ci  = k - tap * Cip;
  float v = (m < O && ci < Cin) ? w[((size_t)m * Cin + ci) * 9 + tap] : 0.0f;
  A[i] = (_Float16)v;
}

__global__ void pack_up(const float* __restrict__ w /*(Ci,O,2,2)*/,
                        _Float16* __restrict__ A, int Ci, int O, int CKp, int Mp)
{
  size_t i = (size_t)blockIdx.x * 256 + threadIdx.x;
  size_t seg = (size_t)Mp * CKp;
  if (i >= 4 * seg) return;
  int rs = (int)(i / seg);
  size_t rem = i - (size_t)rs * seg;
  int m = (int)(rem / CKp);
  int k = (int)(rem - (size_t)m * CKp);
  float v = (m < O && k < Ci) ? w[((size_t)k * O + m) * 4 + rs] : 0.0f;
  A[i] = (_Float16)v;
}

__global__ void bn_prep(const float* __restrict__ bias, const float* __restrict__ g,
                        const float* __restrict__ b, const float* __restrict__ m,
                        const float* __restrict__ v,
                        float* __restrict__ alpha, float* __restrict__ beta, int O)
{
  int i = blockIdx.x * 256 + threadIdx.x;
  if (i >= O) return;
  float a = g[i] * rsqrtf(v[i] + EPSF);
  alpha[i] = a;
  beta[i]  = (bias[i] - m[i]) * a + b[i];
}

__global__ void haar_kernel(const float* __restrict__ X, float* __restrict__ Y,
                            int BC, int H, int W)
{
  size_t i = (size_t)blockIdx.x * 256 + threadIdx.x;
  size_t total = (size_t)BC * H * W;
  if (i >= total) return;
  int w = (int)(i % W);
  size_t t = i / W;
  int h = (int)(t % H);
  size_t bc = t / H;
  Y[i] = 2.0f * X[(bc * (2 * H) + 2 * h) * (2 * W) + 2 * w];
}

__global__ void copy_ch_kernel(const float* __restrict__ S, float* __restrict__ D,
                               int B, int C, int HW, int Ctot, int coff)
{
  size_t i = (size_t)blockIdx.x * 256 + threadIdx.x;
  size_t total = (size_t)B * C * HW;
  if (i >= total) return;
  int hw = (int)(i % HW);
  size_t t = i / HW;
  int c = (int)(t % C);
  int b = (int)(t / C);
  D[((size_t)b * Ctot + coff + c) * HW + hw] = S[i];
}

__global__ void out_conv1x1(const float* __restrict__ X, const float* __restrict__ w,
                            const float* __restrict__ bias, float* __restrict__ Y,
                            int B, int HW)
{
  size_t i = (size_t)blockIdx.x * 256 + threadIdx.x;
  size_t total = (size_t)B * HW;
  if (i >= total) return;
  int hw = (int)(i % HW);
  int b  = (int)(i / HW);
  float s = bias[0];
  #pragma unroll
  for (int c = 0; c < 64; ++c)
    s += w[c] * X[((size_t)b * 64 + c) * HW + hw];
  Y[i] = s;
}

// ---------------------------------------------------------------------------
// Host orchestration
// ---------------------------------------------------------------------------
extern "C" void kernel_launch(void* const* d_in, const int* in_sizes, int n_in,
                              void* d_out, int out_size, void* d_ws, size_t ws_size,
                              hipStream_t stream)
{
  (void)in_sizes; (void)n_in; (void)out_size; (void)ws_size;

  auto P = [&](int i) -> const float* { return (const float*)d_in[i]; };

  struct CBp { const float *b1,*b2,*be1,*g1,*m1,*v1,*be2,*g2,*m2,*v2,*w1,*w2; };
  auto getCB = [&](int base) {
    CBp c;
    c.b1 = P(base + 0);  c.b2 = P(base + 1);
    c.be1 = P(base + 2); c.g1 = P(base + 3); c.m1 = P(base + 4); c.v1 = P(base + 5);
    c.be2 = P(base + 6); c.g2 = P(base + 7); c.m2 = P(base + 8); c.v2 = P(base + 9);
    c.w1 = P(base + 10); c.w2 = P(base + 11);
    return c;
  };

  CBp bott = getCB(0);
  CBp dcb[4]; const float *up_b[4], *up_w[4];
  for (int j = 0; j < 4; ++j) {
    int base = 12 + 14 * j;
    dcb[j] = getCB(base);
    up_b[j] = P(base + 12);
    up_w[j] = P(base + 13);
  }
  CBp ecb[4]; const float *dwt_b[4], *dwt_bnb[4], *dwt_bng[4], *dwt_bnm[4], *dwt_bnv[4], *dwt_w[4];
  for (int j = 0; j < 4; ++j) {
    int base = 68 + 18 * j;
    ecb[j] = getCB(base);
    dwt_b[j]  = P(base + 12);
    dwt_bnb[j] = P(base + 13); dwt_bng[j] = P(base + 14);
    dwt_bnm[j] = P(base + 15); dwt_bnv[j] = P(base + 16);
    dwt_w[j]  = P(base + 17);
  }
  const float* out_b = P(140);
  const float* out_w = P(141);
  const float* x     = P(142);

  char* wsp = (char*)d_ws;
  auto alloc = [&](size_t bytes) -> void* {
    void* p = (void*)wsp;
    wsp += (bytes + 255) & ~(size_t)255;
    return p;
  };
  const size_t BIG = (size_t)4 * 128 * 256 * 256;
  float* P0    = (float*)alloc(BIG * 4);
  float* P1    = (float*)alloc(BIG * 4);
  float* skip0 = (float*)alloc((size_t)4 * 64  * 256 * 256 * 4);
  float* skip1 = (float*)alloc((size_t)4 * 128 * 128 * 128 * 4);
  float* skip2 = (float*)alloc((size_t)4 * 256 * 64  * 64  * 4);
  float* skip3 = (float*)alloc((size_t)4 * 512 * 32  * 32  * 4);
  _Float16* packA = (_Float16*)alloc((size_t)1024 * 9216 * 2);
  float* alphaB = (float*)alloc(1024 * 4);
  float* betaB  = (float*)alloc(1024 * 4);

  const int B_ = 4;

  auto run_conv = [&](const float* X, float* Y, const float* w, const float* bias,
                      const float* g, const float* bb, const float* mm, const float* vv,
                      int Cin, int H, int W, int O, int relu) {
    int Cip = (Cin + 31) & ~31;
    int CKp = 9 * Cip;
    int Mp  = (O + 63) & ~63;
    int sptLog = __builtin_ctz(Cip >> 5);
    size_t pn = (size_t)Mp * CKp;
    pack_w3x3<<<dim3((unsigned)((pn + 255) / 256)), 256, 0, stream>>>(w, packA, O, Cin, Cip, Mp);
    bn_prep<<<dim3((O + 255) / 256), 256, 0, stream>>>(bias, g, bb, mm, vv, alphaB, betaB, O);
    int N = B_ * H * W;
    dim3 grid((N + 63) / 64, Mp / 64, 1);
    if ((Cin & 31) == 0)
      conv3x3_bn_wmma<true><<<grid, 256, 0, stream>>>(packA, CKp, Mp, X, Y, alphaB, betaB,
                                                      Cin, H, W, O, relu, sptLog);
    else
      conv3x3_bn_wmma<false><<<grid, 256, 0, stream>>>(packA, CKp, Mp, X, Y, alphaB, betaB,
                                                       Cin, H, W, O, relu, sptLog);
  };

  auto run_cb = [&](const float* X, float* T, float* Y, const CBp& c,
                    int Cin, int H, int W, int O) {
    run_conv(X, T, c.w1, c.b1, c.g1, c.be1, c.m1, c.v1, Cin, H, W, O, 0);
    run_conv(T, Y, c.w2, c.b2, c.g2, c.be2, c.m2, c.v2, O,   H, W, O, 1);
  };

  auto run_up = [&](const float* X, float* Y, const float* w, const float* bias,
                    int Ci, int H, int W, int O, int Ctot) {
    int CKp = Ci;
    int Mp  = (O + 63) & ~63;
    size_t pn = (size_t)4 * Mp * CKp;
    pack_up<<<dim3((unsigned)((pn + 255) / 256)), 256, 0, stream>>>(w, packA, Ci, O, CKp, Mp);
    int N = B_ * H * W;
    dim3 grid((N + 63) / 64, Mp / 64, 4);
    upconv_wmma<<<grid, 256, 0, stream>>>(packA, CKp, Mp, X, Y, bias,
                                          Ci, H, W, O, Ctot);
  };

  auto run_haar = [&](const float* X, float* Y, int C, int Ho, int Wo) {
    size_t tot = (size_t)B_ * C * Ho * Wo;
    haar_kernel<<<dim3((unsigned)((tot + 255) / 256)), 256, 0, stream>>>(X, Y, B_ * C, Ho, Wo);
  };

  auto run_copy = [&](const float* S, float* D, int C, int HW, int Ctot, int coff) {
    size_t tot = (size_t)B_ * C * HW;
    copy_ch_kernel<<<dim3((unsigned)((tot + 255) / 256)), 256, 0, stream>>>(S, D, B_, C, HW, Ctot, coff);
  };

  auto run_dwt = [&](const float* X, float* Y, int j, int C, int H, int W) {
    run_conv(X, Y, dwt_w[j], dwt_b[j], dwt_bng[j], dwt_bnb[j], dwt_bnm[j], dwt_bnv[j],
             C, H, W, C, 1);
  };

  // ---- encoder -------------------------------------------------------------
  run_cb(x,  P0, skip0, ecb[0],   3, 256, 256,  64);
  run_haar(skip0, P0,  64, 128, 128);
  run_dwt (P0, P1, 0,  64, 128, 128);

  run_cb(P1, P0, skip1, ecb[1],  64, 128, 128, 128);
  run_haar(skip1, P0, 128,  64,  64);
  run_dwt (P0, P1, 1, 128,  64,  64);

  run_cb(P1, P0, skip2, ecb[2], 128,  64,  64, 256);
  run_haar(skip2, P0, 256,  32,  32);
  run_dwt (P0, P1, 2, 256,  32,  32);

  run_cb(P1, P0, skip3, ecb[3], 256,  32,  32, 512);
  run_haar(skip3, P0, 512,  16,  16);
  run_dwt (P0, P1, 3, 512,  16,  16);

  // ---- bottleneck ----------------------------------------------------------
  run_cb(P1, P0, P1, bott, 512, 16, 16, 1024);

  // ---- decoder -------------------------------------------------------------
  run_up(P1, P0, up_w[0], up_b[0], 1024, 16, 16, 512, 1024);
  run_copy(skip3, P0, 512, 32 * 32, 1024, 512);
  run_cb(P0, P1, P0, dcb[0], 1024, 32, 32, 512);

  run_up(P0, P1, up_w[1], up_b[1], 512, 32, 32, 256, 512);
  run_copy(skip2, P1, 256, 64 * 64, 512, 256);
  run_cb(P1, P0, P1, dcb[1], 512, 64, 64, 256);

  run_up(P1, P0, up_w[2], up_b[2], 256, 64, 64, 128, 256);
  run_copy(skip1, P0, 128, 128 * 128, 256, 128);
  run_cb(P0, P1, P0, dcb[2], 256, 128, 128, 128);

  run_up(P0, P1, up_w[3], up_b[3], 128, 128, 128, 64, 128);
  run_copy(skip0, P1, 64, 256 * 256, 128, 64);
  run_cb(P1, P0, P1, dcb[3], 128, 256, 256, 64);

  // ---- output 1x1 conv -----------------------------------------------------
  {
    size_t tot = (size_t)B_ * 256 * 256;
    out_conv1x1<<<dim3((unsigned)((tot + 255) / 256)), 256, 0, stream>>>(
        P1, out_w, out_b, (float*)d_out, B_, 256 * 256);
  }
}
